// Decoder_Step_62586263437395
// MI455X (gfx1250) — matrix operand order repaired and checked by reading, not compiled, not run on hardware
//
#include <hip/hip_runtime.h>
#include <hip/hip_bf16.h>
#include <math.h>

typedef __attribute__((ext_vector_type(16))) __bf16 v16bf;
typedef __attribute__((ext_vector_type(2)))  __bf16 v2bf;
typedef __attribute__((ext_vector_type(8)))  float  v8f;

union FragBF { v16bf v; unsigned u[8]; };
union PackBF { v2bf  v; unsigned u;    };

// ---------------------------------------------------------------------------
// small helpers
// ---------------------------------------------------------------------------
__device__ __forceinline__ float sigm(float x) { return 1.0f / (1.0f + __expf(-x)); }

__device__ __forceinline__ unsigned hash_u32(unsigned x) {
    x ^= x >> 16; x *= 0x7feb352dU;
    x ^= x >> 15; x *= 0x846ca68bU;
    x ^= x >> 16; return x;
}

// pack two f32 into one dword of bf16x2 (lowers to v_cvt_pk_bf16_f32)
__device__ __forceinline__ unsigned pack_bf16x2(float lo, float hi) {
    PackBF t; t.v[0] = (__bf16)lo; t.v[1] = (__bf16)hi; return t.u;
}

// LDS strides (dwords) chosen for bank-conflict-free fragment gathers
#define ASTR 19   // A: 16 k-pairs + pad; odd, 4*inv(19) mod 64 = 44 -> no lo/hi clash
#define BSTR 130  // B: 128 cols + pad; hi-half offset 4*130 mod 64 = 8 -> banks overlap by 8

// ---------------------------------------------------------------------------
// prenet: C = A@W (+bias) [+relu+dropout(p=0.5) if salt != 0]
// tiny (<=15 MFLOP total) -> plain VALU
// ---------------------------------------------------------------------------
__global__ void dense_relu_drop(const float* __restrict__ A,
                                const float* __restrict__ W,
                                const float* __restrict__ bias,
                                float* __restrict__ C,
                                int M, int N, int K, unsigned salt) {
    int idx = blockIdx.x * blockDim.x + threadIdx.x;
    if (idx >= M * N) return;
    int m = idx / N, n = idx % N;
    float acc = bias ? bias[n] : 0.0f;
    for (int k = 0; k < K; ++k)
        acc = fmaf(A[m * K + k], W[k * N + n], acc);
    if (salt) {
        acc = fmaxf(acc, 0.0f);
        unsigned h = hash_u32((unsigned)idx * 2654435761u + salt);
        acc = (h & 1u) ? acc * 2.0f : 0.0f;   // inverted dropout, keep=0.5
    }
    C[idx] = acc;
}

// ---------------------------------------------------------------------------
// score kernel: k = enc @ W_k ; score = v_a . tanh(q + k + b_k) + b_score
// enc viewed as (65536 x 512) f32, W_k (512 x 128) f32.
// 256 thr = 8 waves; block = 128 rows x 128 cols, K chunks of 32,
// LDS tiles pre-packed as bf16x2 dwords.
// ---------------------------------------------------------------------------
__global__ __launch_bounds__(256) void score_wmma(
        const float* __restrict__ enc, const float* __restrict__ Wk,
        const float* __restrict__ bk,  const float* __restrict__ q,
        const float* __restrict__ va,  const float* __restrict__ bscore,
        float* __restrict__ scores) {
    __shared__ unsigned Apk[128 * ASTR];
    __shared__ unsigned Bpk[16 * BSTR];
    const int tid  = threadIdx.x;
    const int lane = tid & 31, w = tid >> 5;
    const int hi   = lane >> 4, lr = lane & 15;
    const int m0   = blockIdx.x * 128;
    const int r19  = (w * 16 + lr) * ASTR;

    v8f zero = {};
    v8f acc[8];
#pragma unroll
    for (int nt = 0; nt < 8; ++nt) acc[nt] = zero;

    for (int kc = 0; kc < 512; kc += 32) {
#pragma unroll
        for (int i = 0; i < 8; ++i) {            // A tile 128x32 -> 128x16 dwords
            int l = tid + i * 256;
            int r = l >> 4, p = l & 15;
            const float2 f2 = *reinterpret_cast<const float2*>(
                enc + (size_t)(m0 + r) * 512 + kc + 2 * p);
            Apk[r * ASTR + p] = pack_bf16x2(f2.x, f2.y);
        }
#pragma unroll
        for (int i = 0; i < 8; ++i) {            // B tile 32x128 -> 16x128 dwords
            int l = tid + i * 256;
            int kp = l >> 7, n = l & 127;
            float lo  = Wk[(size_t)(kc + 2 * kp)     * 128 + n];
            float hiv = Wk[(size_t)(kc + 2 * kp + 1) * 128 + n];
            Bpk[kp * BSTR + n] = pack_bf16x2(lo, hiv);
        }
        __syncthreads();

        FragBF af;
#pragma unroll
        for (int qp = 0; qp < 8; ++qp) {
            int P = qp + 4 * ((qp >> 2) + hi);   // = k/2 per ISA A 16x32 map
            af.u[qp] = Apk[r19 + P];
        }
#pragma unroll
        for (int nt = 0; nt < 8; ++nt) {
            FragBF bf;
#pragma unroll
            for (int qp = 0; qp < 8; ++qp)
                bf.u[qp] = Bpk[(qp + 4 * ((qp >> 2) + hi)) * BSTR + nt * 16 + lr];
            acc[nt] = __builtin_amdgcn_wmma_f32_16x16x32_bf16(
                false, af.v, false, bf.v, (short)0, acc[nt], false, false);
        }
        __syncthreads();
    }

    // fused epilogue: rowsum_j = sum_n va[n]*tanh(q[b,n] + bk[n] + k[row,n])
    const int b = m0 >> 9;                       // 128 rows always within one b
    float rowsum[8];
#pragma unroll
    for (int j = 0; j < 8; ++j) rowsum[j] = 0.0f;
#pragma unroll
    for (int nt = 0; nt < 8; ++nt) {
        int n = nt * 16 + lr;
        float qv  = q[b * 128 + n];
        float bkv = bk[n];
        float vav = va[n];
#pragma unroll
        for (int j = 0; j < 8; ++j)
            rowsum[j] += vav * tanhf(qv + bkv + acc[nt][j]);
    }
#pragma unroll
    for (int j = 0; j < 8; ++j) {
        float s = rowsum[j];
        s += __shfl_xor(s, 1);
        s += __shfl_xor(s, 2);
        s += __shfl_xor(s, 4);
        s += __shfl_xor(s, 8);                   // reduce within 16-lane half
        if (lr == 0)
            scores[m0 + w * 16 + j + 8 * hi] = s + bscore[0];
    }
}

// ---------------------------------------------------------------------------
// monotonic alignment scan (sequential over T per batch row)
// ---------------------------------------------------------------------------
__global__ void align_scan(const float* __restrict__ scores,
                           const float* __restrict__ prev_align,
                           float* __restrict__ align_out) {
    int b = blockIdx.x * blockDim.x + threadIdx.x;
    if (b >= 128) return;
    float run_prod = 1.0f, run_sum = 0.0f;
    for (int t = 0; t < 512; ++t) {
        float p  = sigm(scores[b * 512 + t]);
        float cp = fminf(fmaxf(run_prod, 1e-10f), 1.0f);
        run_sum += prev_align[b * 512 + t] / cp;
        align_out[b * 512 + t] = p * run_prod * run_sum;
        run_prod *= (1.0f - p);
    }
}

// ---------------------------------------------------------------------------
// attentions[b,v] = sum_t align[b,t] * enc[b,t,v]  (streaming, coalesced in v)
// ---------------------------------------------------------------------------
__global__ __launch_bounds__(256) void attn_kernel(const float* __restrict__ enc,
                                                   const float* __restrict__ align,
                                                   float* __restrict__ att) {
    __shared__ float al[512];
    int b = blockIdx.y;
    int v = blockIdx.x * 256 + threadIdx.x;
    al[threadIdx.x]       = align[b * 512 + threadIdx.x];
    al[256 + threadIdx.x] = align[b * 512 + 256 + threadIdx.x];
    __syncthreads();
    float acc = 0.0f;
    const float* e = enc + (size_t)b * 512 * 512 + v;
    for (int t = 0; t < 512; ++t)
        acc = fmaf(al[t], e[(size_t)t * 512], acc);
    att[b * 512 + v] = acc;
}

// ---------------------------------------------------------------------------
// generic bf16 WMMA GEMM: C[128*gridY x N] (+)= A[.. x K] @ B[K x N] (+bias)
// K multiple of 32; N guarded. 256 thr, 128x128 tile per block.
// ---------------------------------------------------------------------------
__global__ __launch_bounds__(256) void gemm_wmma_bf16(
        const float* __restrict__ A, int lda,
        const float* __restrict__ B, int ldb,
        float* __restrict__ C, int ldc,
        int N, int K, const float* __restrict__ bias, int accumulate) {
    __shared__ unsigned Apk[128 * ASTR];
    __shared__ unsigned Bpk[16 * BSTR];
    const int tid  = threadIdx.x;
    const int lane = tid & 31, w = tid >> 5;
    const int hi   = lane >> 4, lr = lane & 15;
    const int m0   = blockIdx.y * 128;
    const int n0   = blockIdx.x * 128;
    const int r19  = (w * 16 + lr) * ASTR;

    v8f zero = {};
    v8f acc[8];
#pragma unroll
    for (int nt = 0; nt < 8; ++nt) acc[nt] = zero;

    for (int kc = 0; kc < K; kc += 32) {
#pragma unroll
        for (int i = 0; i < 8; ++i) {            // A tile 128x32 -> packed
            int l = tid + i * 256;
            int r = l >> 4, p = l & 15;
            const float2 f2 = *reinterpret_cast<const float2*>(
                A + (size_t)(m0 + r) * lda + kc + 2 * p);
            Apk[r * ASTR + p] = pack_bf16x2(f2.x, f2.y);
        }
#pragma unroll
        for (int i = 0; i < 8; ++i) {            // B tile 32x128 -> packed (N guard)
            int l = tid + i * 256;
            int kp = l >> 7, n = l & 127;
            float lo = 0.0f, hiv = 0.0f;
            if (n0 + n < N) {
                lo  = B[(size_t)(kc + 2 * kp)     * ldb + n0 + n];
                hiv = B[(size_t)(kc + 2 * kp + 1) * ldb + n0 + n];
            }
            Bpk[kp * BSTR + n] = pack_bf16x2(lo, hiv);
        }
        __syncthreads();

        FragBF af;
#pragma unroll
        for (int qp = 0; qp < 8; ++qp) {
            int P = qp + 4 * ((qp >> 2) + hi);
            af.u[qp] = Apk[r19 + P];
        }
#pragma unroll
        for (int nt = 0; nt < 8; ++nt) {
            FragBF bf;
#pragma unroll
            for (int qp = 0; qp < 8; ++qp)
                bf.u[qp] = Bpk[(qp + 4 * ((qp >> 2) + hi)) * BSTR + nt * 16 + lr];
            acc[nt] = __builtin_amdgcn_wmma_f32_16x16x32_bf16(
                false, af.v, false, bf.v, (short)0, acc[nt], false, false);
        }
        __syncthreads();
    }

    const int row_base = m0 + w * 16;
#pragma unroll
    for (int nt = 0; nt < 8; ++nt) {
        int col = n0 + nt * 16 + lr;
        if (col < N) {
#pragma unroll
            for (int j = 0; j < 8; ++j) {
                int row = row_base + j + 8 * hi;
                size_t o = (size_t)row * ldc + col;
                float v = acc[nt][j];
                if (accumulate) C[o] += v;
                else            C[o] = v + (bias ? bias[col] : 0.0f);
            }
        }
    }
}

// ---------------------------------------------------------------------------
// LSTM gates: z = [i|f|g|o] blocks of 1024; c' = sig(f)c + sig(i)tanh(g)
// ---------------------------------------------------------------------------
__global__ void lstm_gates(const float* __restrict__ z,
                           const float* __restrict__ c_old,
                           float* __restrict__ h_out, float* __restrict__ c_out) {
    int idx = blockIdx.x * blockDim.x + threadIdx.x;
    if (idx >= 128 * 1024) return;
    int b = idx >> 10, r = idx & 1023;
    const float* zb = z + (size_t)b * 4096;
    float gi = zb[r], gf = zb[1024 + r], gg = zb[2048 + r], go = zb[3072 + r];
    float cn = sigm(gf) * c_old[idx] + sigm(gi) * tanhf(gg);
    h_out[idx] = sigm(go) * tanhf(cn);
    c_out[idx] = cn;
}

// concat along feature dim: C[128 x (na+nb)] = [A | B]
__global__ void concat2(const float* __restrict__ A, int na,
                        const float* __restrict__ B, int nb,
                        float* __restrict__ C) {
    int idx = blockIdx.x * blockDim.x + threadIdx.x;
    int n = na + nb;
    if (idx >= 128 * n) return;
    int m = idx / n, c = idx % n;
    C[idx] = (c < na) ? A[m * na + c] : B[m * nb + (c - na)];
}

__global__ void split_proj(const float* __restrict__ proj,
                           float* __restrict__ mel, float* __restrict__ stops) {
    int idx = blockIdx.x * blockDim.x + threadIdx.x;
    if (idx >= 128 * 81) return;
    int m = idx / 81, c = idx % 81;
    float v = proj[idx];
    if (c < 80) mel[m * 80 + c] = v;
    else        stops[m] = v;
}

// ---------------------------------------------------------------------------
extern "C" void kernel_launch(void* const* d_in, const int* in_sizes, int n_in,
                              void* d_out, int out_size, void* d_ws, size_t ws_size,
                              hipStream_t stream) {
    const float* enc        = (const float*)d_in[0];
    const float* mels       = (const float*)d_in[1];
    const float* prev_align = (const float*)d_in[2];
    const float* h1         = (const float*)d_in[3];
    const float* c1         = (const float*)d_in[4];
    const float* h2         = (const float*)d_in[5];
    const float* c2         = (const float*)d_in[6];
    const float* W_pre1     = (const float*)d_in[7];
    const float* b_pre1     = (const float*)d_in[8];
    const float* W_pre2     = (const float*)d_in[9];
    const float* b_pre2     = (const float*)d_in[10];
    const float* W_q        = (const float*)d_in[11];
    const float* W_k        = (const float*)d_in[12];
    const float* b_k        = (const float*)d_in[13];
    const float* v_a        = (const float*)d_in[14];
    const float* b_score    = (const float*)d_in[15];
    const float* Wx1        = (const float*)d_in[16];
    const float* Wh1        = (const float*)d_in[17];
    const float* b_l1       = (const float*)d_in[18];
    const float* Wx2        = (const float*)d_in[19];
    const float* Wh2        = (const float*)d_in[20];
    const float* b_l2       = (const float*)d_in[21];
    const float* W_proj     = (const float*)d_in[22];
    const float* b_proj     = (const float*)d_in[23];

    float* out       = (float*)d_out;
    float* mel_out   = out;                 // 128*80
    float* stops     = out + 10240;         // 128
    float* align_out = out + 10368;         // 128*512
    float* h1n       = out + 75904;         // 128*1024
    float* c1n       = out + 206976;
    float* h2n       = out + 338048;
    float* c2n       = out + 469120;

    float* ws      = (float*)d_ws;
    float* x1      = ws;                    // 128*256
    float* x       = ws + 32768;            // 128*256
    float* q       = ws + 65536;            // 128*128
    float* scores  = ws + 81920;            // 128*512
    float* att     = ws + 147456;           // 128*512
    float* rnn_in  = ws + 212992;           // 128*768
    float* z       = ws + 311296;           // 128*4096 (reused for both LSTMs)
    float* proj_in = ws + 835584;           // 128*1536
    float* proj    = ws + 1032192;          // 128*81

    dim3 blk(256);

    // prenet + query
    dense_relu_drop<<<128, blk, 0, stream>>>(mels, W_pre1, b_pre1, x1, 128, 256, 80,  0x1234567u);
    dense_relu_drop<<<128, blk, 0, stream>>>(x1,   W_pre2, b_pre2, x,  128, 256, 256, 0x89abcdeu);
    dense_relu_drop<<<64,  blk, 0, stream>>>(x,    W_q,    nullptr, q, 128, 128, 256, 0u);

    // attention energies (big WMMA GEMM fused with tanh/dot epilogue)
    score_wmma<<<512, blk, 0, stream>>>(enc, W_k, b_k, q, v_a, b_score, scores);

    // monotonic alignment scan
    align_scan<<<1, 128, 0, stream>>>(scores, prev_align, align_out);

    // context vectors (second streaming pass over encodings; L2-resident)
    attn_kernel<<<dim3(2, 128), blk, 0, stream>>>(enc, align_out, att);

    // LSTM 1
    concat2<<<384, blk, 0, stream>>>(x, 256, att, 512, rnn_in);
    gemm_wmma_bf16<<<dim3(32, 1), blk, 0, stream>>>(rnn_in, 768,  Wx1, 4096, z, 4096, 4096, 768,  b_l1, 0);
    gemm_wmma_bf16<<<dim3(32, 1), blk, 0, stream>>>(h1,     1024, Wh1, 4096, z, 4096, 4096, 1024, nullptr, 1);
    lstm_gates<<<512, blk, 0, stream>>>(z, c1, h1n, c1n);

    // LSTM 2
    gemm_wmma_bf16<<<dim3(32, 1), blk, 0, stream>>>(h1n, 1024, Wx2, 4096, z, 4096, 4096, 1024, b_l2, 0);
    gemm_wmma_bf16<<<dim3(32, 1), blk, 0, stream>>>(h2,  1024, Wh2, 4096, z, 4096, 4096, 1024, nullptr, 1);
    lstm_gates<<<512, blk, 0, stream>>>(z, c2, h2n, c2n);

    // projection
    concat2<<<768, blk, 0, stream>>>(h2n, 1024, att, 512, proj_in);
    gemm_wmma_bf16<<<dim3(1, 1), blk, 0, stream>>>(proj_in, 1536, W_proj, 81, proj, 81, 81, 1536, b_proj, 0);
    split_proj<<<41, blk, 0, stream>>>(proj, mel_out, stops);
}
